// LSTMAE_79791902425463
// MI455X (gfx1250) — matrix operand order repaired and checked by reading, not compiled, not run on hardware
//
#include <hip/hip_runtime.h>
#include <hip/hip_bf16.h>

typedef __attribute__((ext_vector_type(16))) __bf16 v16bf;
typedef __attribute__((ext_vector_type(8)))  float  v8f;
typedef __attribute__((ext_vector_type(4)))  unsigned int v4u;

#define ASTR 520          // LDS A-buffer row stride in halves (1040B: 16B-aligned, bank-rotating)
#define H_DIM 256
#define T_STEPS 128
#define BTILE 16

union FragU { v4u q[2]; v16bf v; };

__device__ __forceinline__ unsigned short f2bf(float f) {
    unsigned u = __builtin_bit_cast(unsigned, f);
    u += 0x7FFFu + ((u >> 16) & 1u);          // round-to-nearest-even
    return (unsigned short)(u >> 16);
}

// Fast transcendentals: v_rcp_f32 instead of the IEEE divide expansion — the elementwise
// phase sits on the sequential critical path between GEMMs 512 times.
__device__ __forceinline__ float rcp_fast(float x) { return __builtin_amdgcn_rcpf(x); }
__device__ __forceinline__ float sigf(float x) {
    return rcp_fast(1.0f + __expf(-x));
}
__device__ __forceinline__ float tanh_fast(float x) {
    x = fminf(fmaxf(x, -15.0f), 15.0f);
    float e = __expf(2.0f * x);
    return (e - 1.0f) * rcp_fast(e + 1.0f);
}

// A fragment (16x32 bf16): lane holds row m=lane&15; lo-half lanes K {kk..kk+7, kk+16..kk+23},
// hi-half lanes K {kk+8..kk+15, kk+24..kk+31}  -> two ds_load_b128.
__device__ __forceinline__ v16bf lda(const unsigned short* A, int row, int col) {
    FragU u;
    u.q[0] = *(const v4u*)(A + row * ASTR + col);
    u.q[1] = *(const v4u*)(A + row * ASTR + col + 16);
    return u.v;
}
// B fragment: packed weights are fragment-contiguous: 512 halves per (ntile,kchunk) block,
// 16 halves per lane -> two coalesced global b128 loads.
__device__ __forceinline__ v16bf ldb(const unsigned short* __restrict__ p) {
    FragU u;
    u.q[0] = *(const v4u*)(p);
    u.q[1] = *(const v4u*)(p + 8);
    return u.v;
}

// Gates GEMM: acc[g2][q] = bias + A[16 x Kc] * W[Kc x 1024].
// Wave owns hblocks hb0,hb1; ntile(q,hb) = q*16+hb so i/f/g/o for the same h columns land in
// one wave's registers. K loop NOT unrolled (register pressure): per iteration, batch-load the
// 8 B fragments (16 coalesced b128s in flight), prefetch the next K-chunk, then 8 WMMAs.
template <int KCH>
__device__ __forceinline__ void gemm_gates(const unsigned short* __restrict__ Wp,
                                           const float bv[2][4],
                                           const unsigned short* A,
                                           int hb0, int hb1, int al, int abase, int lane,
                                           v8f acc[2][4]) {
#pragma unroll
    for (int g2 = 0; g2 < 2; ++g2)
#pragma unroll
        for (int q = 0; q < 4; ++q)
#pragma unroll
            for (int j = 0; j < 8; ++j) acc[g2][q][j] = bv[g2][q];

    const unsigned short* wlane = Wp + (lane << 4);    // per-lane base inside each 512-half block
#pragma unroll 1
    for (int kc = 0; kc < KCH; ++kc) {
        v16bf a = lda(A, al, (kc << 5) + abase);
        v16bf bf[2][4];
#pragma unroll
        for (int g2 = 0; g2 < 2; ++g2) {
            int hb = g2 ? hb1 : hb0;
#pragma unroll
            for (int q = 0; q < 4; ++q) {
                int nt = (q << 4) + hb;
                bf[g2][q] = ldb(wlane + ((size_t)(nt * KCH + kc) << 9));
                __builtin_prefetch((const void*)(wlane + ((size_t)(nt * KCH + kc + 1) << 9)), 0, 3);
            }
        }
#pragma unroll
        for (int g2 = 0; g2 < 2; ++g2)
#pragma unroll
            for (int q = 0; q < 4; ++q)
                acc[g2][q] = __builtin_amdgcn_wmma_f32_16x16x32_bf16(
                    false, a, false, bf[g2][q], (short)0, acc[g2][q], false, false);
    }
}

// LSTM elementwise: c in registers, write bf16 h into A buffers for the next GEMM.
__device__ __forceinline__ void lstm_ew(v8f acc[2][4], float c[2][8],
                                        unsigned short* dst1, int off1,
                                        unsigned short* dst2, int off2,
                                        int hb0, int hb1, int al, int hi) {
#pragma unroll
    for (int g2 = 0; g2 < 2; ++g2) {
        int hb = g2 ? hb1 : hb0;
        int hcol = (hb << 4) + al;
#pragma unroll
        for (int j = 0; j < 8; ++j) {
            int m = j + (hi << 3);
            float iv = acc[g2][0][j], fv = acc[g2][1][j];
            float gv = acc[g2][2][j], ov = acc[g2][3][j];
            float nc = sigf(fv) * c[g2][j] + sigf(iv) * tanh_fast(gv);
            c[g2][j] = nc;
            unsigned short hb16 = f2bf(sigf(ov) * tanh_fast(nc));
            dst1[m * ASTR + off1 + hcol] = hb16;
            if (dst2) dst2[m * ASTR + off2 + hcol] = hb16;
        }
    }
}

__global__ __launch_bounds__(256, 1) void lstmae_main(
    const float* __restrict__ x,
    const unsigned short* __restrict__ W0E, const unsigned short* __restrict__ W1E,
    const unsigned short* __restrict__ W0D, const unsigned short* __restrict__ W1D,
    const unsigned short* __restrict__ WP,
    const float* __restrict__ b0e, const float* __restrict__ b1e,
    const float* __restrict__ b0d, const float* __restrict__ b1d,
    const float* __restrict__ bp,
    float* __restrict__ out) {
    __shared__ unsigned short A0[BTILE * ASTR];   // enc: [x|h0] K=320 ; dec: [z|h_d0] K=512
    __shared__ unsigned short A1[BTILE * ASTR];   // enc: [h0|h1]     ; dec: [h_d0|h_d1]

    const int tid  = threadIdx.x;
    const int lane = tid & 31;
    const int w    = tid >> 5;         // 8 waves
    const int al   = lane & 15;
    const int hi   = lane >> 4;
    const int abase = hi << 3;
    const int hb0 = w, hb1 = w + 8;
    const int b0 = blockIdx.x * BTILE;

    // Hoist wave-invariant bias values into registers (reused for all 256 timesteps).
    float bv0e[2][4], bv1e[2][4], bv0d[2][4], bv1d[2][4];
#pragma unroll
    for (int g2 = 0; g2 < 2; ++g2) {
        int hb = g2 ? hb1 : hb0;
#pragma unroll
        for (int q = 0; q < 4; ++q) {
            int idx = (q << 8) + (hb << 4) + al;
            bv0e[g2][q] = b0e[idx];
            bv1e[g2][q] = b1e[idx];
            bv0d[g2][q] = b0d[idx];
            bv1d[g2][q] = b1d[idx];
        }
    }
    const float bpv = (w < 3) ? bp[(w << 4) + al] : 0.f;

    for (int i = tid; i < BTILE * ASTR; i += 256) { A0[i] = 0; A1[i] = 0; }
    float c0[2][8], c1[2][8];
#pragma unroll
    for (int g2 = 0; g2 < 2; ++g2)
#pragma unroll
        for (int j = 0; j < 8; ++j) { c0[g2][j] = 0.f; c1[g2][j] = 0.f; }
    __syncthreads();

    v8f acc[2][4];

    // ---------------- Encoder: 128 steps, 2 fused layers ----------------
    for (int t = 0; t < T_STEPS; ++t) {
        for (int i = tid; i < BTILE * 38; i += 256) {        // stage x_t (cols 0..37; 38..63 stay 0)
            int r = i / 38, cx = i - r * 38;
            A0[r * ASTR + cx] = f2bf(x[((size_t)(b0 + r) * T_STEPS + t) * 38 + cx]);
        }
        __syncthreads();
        gemm_gates<10>(W0E, bv0e, A0, hb0, hb1, al, abase, lane, acc);  // K=320: [x|h0]
        __syncthreads();
        lstm_ew(acc, c0, A0, 64, A1, 0, hb0, hb1, al, hi);              // h0 -> A0[64..] & A1[0..]
        __syncthreads();
        gemm_gates<16>(W1E, bv1e, A1, hb0, hb1, al, abase, lane, acc);  // K=512: [h0|h1]
        __syncthreads();
        lstm_ew(acc, c1, A1, 256, nullptr, 0, hb0, hb1, al, hi);        // h1 -> A1[256..]
        __syncthreads();
    }

    // ---------------- Latent -> decoder prep ----------------
    for (int i = tid; i < BTILE * H_DIM; i += 256) {   // z = h1 into A0[0..255]; clear A0[256..]
        int r = i >> 8, cx = i & 255;
        A0[r * ASTR + cx] = A1[r * ASTR + 256 + cx];
        A0[r * ASTR + 256 + cx] = 0;
    }
    __syncthreads();
    for (int i = tid; i < BTILE * 512; i += 256) { int r = i >> 9, cx = i & 511; A1[r * ASTR + cx] = 0; }
#pragma unroll
    for (int g2 = 0; g2 < 2; ++g2)
#pragma unroll
        for (int j = 0; j < 8; ++j) { c0[g2][j] = 0.f; c1[g2][j] = 0.f; }
    __syncthreads();

    // ---------------- Decoder: 128 steps, 2 layers + fused projection ----------------
    for (int t = 0; t < T_STEPS; ++t) {
        gemm_gates<16>(W0D, bv0d, A0, hb0, hb1, al, abase, lane, acc);  // K=512: [z|h_d0]
        __syncthreads();
        lstm_ew(acc, c0, A0, 256, A1, 0, hb0, hb1, al, hi);             // h_d0
        __syncthreads();
        gemm_gates<16>(W1D, bv1d, A1, hb0, hb1, al, abase, lane, acc);  // K=512: [h_d0|h_d1]
        __syncthreads();
        lstm_ew(acc, c1, A1, 256, nullptr, 0, hb0, hb1, al, hi);        // h_d1 -> A1[256..]
        __syncthreads();
        if (w < 3) {                                                    // proj: [16,256]x[256,48]
            int n = (w << 4) + al;
            const unsigned short* wplane = WP + (lane << 4);
            v8f pa;
#pragma unroll
            for (int j = 0; j < 8; ++j) pa[j] = bpv;
#pragma unroll 2
            for (int kc = 0; kc < 8; ++kc) {
                v16bf a = lda(A1, al, 256 + (kc << 5) + abase);
                v16bf b = ldb(wplane + ((size_t)(w * 8 + kc) << 9));
                pa = __builtin_amdgcn_wmma_f32_16x16x32_bf16(false, a, false, b, (short)0, pa, false, false);
            }
            if (n < 38) {
#pragma unroll
                for (int j = 0; j < 8; ++j) {
                    int m = j + (hi << 3);
                    out[((size_t)(b0 + m) * T_STEPS + t) * 38 + n] = pa[j];
                }
            }
        }
    }
}

// Pack f32 weights into bf16, B-fragment-contiguous layout:
// block(nt,kc) = 512 halves; lane holds n=nt*16+(lane&15), K = kc*32 + (lane>=16?8:0) + {0..7, 16..23}.
// Combined K axis: [input-weight region (wihPad, zero-padded past wihCols) | Whh (256)].
__global__ void pack_w(unsigned short* __restrict__ dst,
                       const float* __restrict__ wih, int wihCols, int wihPad,
                       const float* __restrict__ whh, int Kc, int nrows, int count) {
    int kch = Kc >> 5;
    for (int e = blockIdx.x * blockDim.x + threadIdx.x; e < count; e += gridDim.x * blockDim.x) {
        int j = e & 15, fid = e >> 4;
        int lane = fid & 31, blk = fid >> 5;
        int kc = blk % kch, nt = blk / kch;
        int n = (nt << 4) + (lane & 15);
        int k = (kc << 5) + ((lane >> 4) << 3) + (j & 7) + ((j >> 3) << 4);
        float v = 0.f;
        if (n < nrows) {
            if (k < wihPad) { if (k < wihCols) v = wih[n * wihCols + k]; }
            else            v = whh[n * 256 + (k - wihPad)];
        }
        dst[e] = f2bf(v);
    }
}

__global__ void pack_bias(float* __restrict__ dst, const float* __restrict__ a,
                          const float* __restrict__ b, int n) {
    int i = blockIdx.x * blockDim.x + threadIdx.x;
    if (i < n) dst[i] = a[i] + b[i];
}
__global__ void pack_bias_pad(float* __restrict__ dst, const float* __restrict__ a, int n, int total) {
    int i = blockIdx.x * blockDim.x + threadIdx.x;
    if (i < total) dst[i] = (i < n) ? a[i] : 0.f;
}

extern "C" void kernel_launch(void* const* d_in, const int* in_sizes, int n_in,
                              void* d_out, int out_size, void* d_ws, size_t ws_size,
                              hipStream_t stream) {
    (void)in_sizes; (void)n_in; (void)out_size; (void)ws_size;
    const float* x        = (const float*)d_in[0];
    const float* eWih0    = (const float*)d_in[1];
    const float* eWhh0    = (const float*)d_in[2];
    const float* eBih0    = (const float*)d_in[3];
    const float* eBhh0    = (const float*)d_in[4];
    const float* eWih1    = (const float*)d_in[5];
    const float* eWhh1    = (const float*)d_in[6];
    const float* eBih1    = (const float*)d_in[7];
    const float* eBhh1    = (const float*)d_in[8];
    const float* dWih0    = (const float*)d_in[9];
    const float* dWhh0    = (const float*)d_in[10];
    const float* dBih0    = (const float*)d_in[11];
    const float* dBhh0    = (const float*)d_in[12];
    const float* dWih1    = (const float*)d_in[13];
    const float* dWhh1    = (const float*)d_in[14];
    const float* dBih1    = (const float*)d_in[15];
    const float* dBhh1    = (const float*)d_in[16];
    const float* projW    = (const float*)d_in[17];
    const float* projB    = (const float*)d_in[18];

    // Workspace layout (halves): W0E[1024*320] W1E[1024*512] W0D[1024*512] W1D[1024*512] WP[48*256], then f32 biases.
    unsigned short* ws16 = (unsigned short*)d_ws;
    unsigned short* W0E = ws16;
    unsigned short* W1E = W0E + 327680;
    unsigned short* W0D = W1E + 524288;
    unsigned short* W1D = W0D + 524288;
    unsigned short* WP  = W1D + 524288;      // 12288 halves
    float* bias = (float*)((char*)d_ws + 3825664);
    float* b0e = bias, *b1e = bias + 1024, *b0d = bias + 2048, *b1d = bias + 3072, *bp = bias + 4096;

    pack_w<<<(327680 + 255) / 256, 256, 0, stream>>>(W0E, eWih0, 38, 64, eWhh0, 320, 1024, 327680);
    pack_w<<<(524288 + 255) / 256, 256, 0, stream>>>(W1E, eWih1, 256, 256, eWhh1, 512, 1024, 524288);
    pack_w<<<(524288 + 255) / 256, 256, 0, stream>>>(W0D, dWih0, 256, 256, dWhh0, 512, 1024, 524288);
    pack_w<<<(524288 + 255) / 256, 256, 0, stream>>>(W1D, dWih1, 256, 256, dWhh1, 512, 1024, 524288);
    pack_w<<<(12288 + 255) / 256, 256, 0, stream>>>(WP, projW, 256, 256, projW, 256, 38, 12288);
    pack_bias<<<4, 256, 0, stream>>>(b0e, eBih0, eBhh0, 1024);
    pack_bias<<<4, 256, 0, stream>>>(b1e, eBih1, eBhh1, 1024);
    pack_bias<<<4, 256, 0, stream>>>(b0d, dBih0, dBhh0, 1024);
    pack_bias<<<4, 256, 0, stream>>>(b1d, dBih1, dBhh1, 1024);
    pack_bias_pad<<<1, 64, 0, stream>>>(bp, projB, 38, 48);

    lstmae_main<<<64, 256, 0, stream>>>(x, W0E, W1E, W0D, W1D, WP,
                                        b0e, b1e, b0d, b1d, bp, (float*)d_out);
}
